// KiloNeRF_7129645711615
// MI455X (gfx1250) — matrix operand order repaired and checked by reading, not compiled, not run on hardware
//
#include <hip/hip_runtime.h>

#define NGRID   16
#define NCELLS  4096
#define SCALEF  3.0f

typedef __attribute__((ext_vector_type(16))) _Float16 v16h;
typedef __attribute__((ext_vector_type(8)))  float    v8f;

union FragU { v16h v; unsigned u[8]; _Float16 h[16]; };

// Per-wave LDS ordering fence (CDNA5 split counter). LDS ops are in-order per
// wave; the asm + memory clobber also stops the compiler reordering LDS traffic.
#define WAVE_LDS_FENCE() asm volatile("s_wait_dscnt 0" ::: "memory")

__device__ __forceinline__ int clamp16(int v) { return v < 0 ? 0 : (v > 15 ? 15 : v); }

__device__ __forceinline__ int cell_of(float x0, float x1, float x2) {
  const float invh = (float)NGRID / SCALEF;  // x / (SCALE/N)
  int i0 = clamp16((int)(x0 * invh + 8.0f)); // trunc-toward-zero like astype(int32)
  int i1 = clamp16((int)(x1 * invh + 8.0f));
  int i2 = clamp16((int)(x2 * invh + 8.0f));
  return (i0 * 16 + i1) * 16 + i2;
}

// ---- WMMA fragment builders --------------------------------------------------
// A (16x32 f16): lane l -> row M=l&15; VGPR v0..3 hold K=(l>>4)*8+2v,(+1);
//                v4..7 hold K=16+(l>>4)*8+2(v-4),(+1).  Staging is [m][Kpad] f16.
__device__ __forceinline__ v16h load_a_frag(const _Float16* A, int lane, int kt) {
  FragU f;
  const int m = lane & 15, hs = lane >> 4;
  const unsigned* row = (const unsigned*)(A + m * 64);
  const int b0 = (kt * 32 + hs * 8) >> 1;
  const int b1 = (kt * 32 + 16 + hs * 8) >> 1;
#pragma unroll
  for (int v = 0; v < 4; ++v) f.u[v] = row[b0 + v];
#pragma unroll
  for (int v = 0; v < 4; ++v) f.u[4 + v] = row[b1 + v];
  return f.v;
}

// B (32x16 f16): lane l -> col N=l&15; VGPR v0..3 hold K=(l>>4)*8+2v,(+1);
//                v4..7 hold K=16+(l>>4)*8+2(v-4),(+1).
// Weights stored transposed in LDS as WT[n][k] (kstride halves per row).
__device__ __forceinline__ v16h load_b_frag(const _Float16* WT, int kstride,
                                            int lane, int kt, int nt) {
  FragU f;
  const int n = nt * 16 + (lane & 15);
  const unsigned* row = (const unsigned*)(WT + n * kstride);
  const int b0 = (kt * 32 + (lane >> 4) * 8) >> 1;
  const int b1 = (kt * 32 + 16 + (lane >> 4) * 8) >> 1;
#pragma unroll
  for (int v = 0; v < 4; ++v) f.u[v] = row[b0 + v];
#pragma unroll
  for (int v = 0; v < 4; ++v) f.u[4 + v] = row[b1 + v];
  return f.v;
}

#define WMMA(a, b, c) __builtin_amdgcn_wmma_f32_16x16x32_f16(false, (a), false, (b), (short)0, (c), false, false)

// ---- Binning kernels ---------------------------------------------------------
__global__ void count_kernel(const float* __restrict__ x, int* __restrict__ counts, int P) {
  int p = blockIdx.x * 256 + threadIdx.x;
  if (p >= P) return;
  atomicAdd(&counts[cell_of(x[3 * p], x[3 * p + 1], x[3 * p + 2])], 1);
}

__global__ void scan_kernel(const int* __restrict__ counts, int* __restrict__ offsets,
                            int* __restrict__ cursor) {
  __shared__ int s[1024];
  const int t = threadIdx.x, base = t * 4;
  int c0 = counts[base], c1 = counts[base + 1], c2 = counts[base + 2], c3 = counts[base + 3];
  int sum = c0 + c1 + c2 + c3;
  s[t] = sum;
  __syncthreads();
  for (int off = 1; off < 1024; off <<= 1) {
    int v = (t >= off) ? s[t - off] : 0;
    __syncthreads();
    s[t] += v;
    __syncthreads();
  }
  int excl = s[t] - sum;
  offsets[base] = excl;          cursor[base] = excl;
  offsets[base + 1] = excl + c0; cursor[base + 1] = excl + c0;
  offsets[base + 2] = excl + c0 + c1; cursor[base + 2] = excl + c0 + c1;
  offsets[base + 3] = excl + c0 + c1 + c2; cursor[base + 3] = excl + c0 + c1 + c2;
  if (t == 1023) offsets[4096] = excl + sum;
}

__global__ void scatter_kernel(const float* __restrict__ x, int* __restrict__ cursor,
                               int* __restrict__ pidx, int P) {
  int p = blockIdx.x * 256 + threadIdx.x;
  if (p >= P) return;
  int c = cell_of(x[3 * p], x[3 * p + 1], x[3 * p + 2]);
  pidx[atomicAdd(&cursor[c], 1)] = p;
}

// ---- Per-cell fused MLP (one block per cell; 8 waves; 16 points per WMMA tile)
__global__ __launch_bounds__(256) void mlp_kernel(
    const float* __restrict__ xin, const float* __restrict__ din,
    const float* __restrict__ w1g, const float* __restrict__ b1g,
    const float* __restrict__ w2g, const float* __restrict__ b2g,
    const float* __restrict__ w3g, const float* __restrict__ b3g,
    const float* __restrict__ w4g, const float* __restrict__ b4g,
    const float* __restrict__ w5g, const float* __restrict__ b5g,
    const int* __restrict__ offsets, const int* __restrict__ pidx,
    float* __restrict__ outC, float* __restrict__ outD) {
  // Weights, transposed [n][k] f16, zero padded
  __shared__ _Float16 w1T[32 * 64];  // 63->64 K, N=32
  __shared__ _Float16 w2T[48 * 32];  // K=32, 33->48 N
  __shared__ _Float16 w3T[32 * 32];
  __shared__ _Float16 w4T[32 * 64];  // 59->64 K, N=32
  __shared__ _Float16 w5T[16 * 32];  // K=32, 3->16 N
  __shared__ float bias1[32], bias2[48], bias3[32], bias4[32], bias5[16];
  // Per-wave activation staging [16][64] + dir-encoding [16][32] + row metadata
  __shared__ _Float16 stagAll[8 * 16 * 64];
  __shared__ _Float16 edAll[8 * 16 * 32];
  __shared__ float sigAll[8 * 16];
  __shared__ int pmAll[8 * 16];
  __shared__ int mkAll[8 * 16];

  const int cell = blockIdx.x;
  const int tid = threadIdx.x;

  {  // stream this cell's weights from HBM exactly once -> f16 in LDS
    const float* g1 = w1g + (size_t)cell * 63 * 32;
    for (int i = tid; i < 32 * 64; i += 256) { int n = i >> 6, k = i & 63; w1T[i] = (_Float16)((k < 63) ? g1[k * 32 + n] : 0.f); }
    const float* g2 = w2g + (size_t)cell * 32 * 33;
    for (int i = tid; i < 48 * 32; i += 256) { int n = i >> 5, k = i & 31; w2T[i] = (_Float16)((n < 33) ? g2[k * 33 + n] : 0.f); }
    const float* g3 = w3g + (size_t)cell * 32 * 32;
    for (int i = tid; i < 32 * 32; i += 256) { int n = i >> 5, k = i & 31; w3T[i] = (_Float16)g3[k * 32 + n]; }
    const float* g4 = w4g + (size_t)cell * 59 * 32;
    for (int i = tid; i < 32 * 64; i += 256) { int n = i >> 6, k = i & 63; w4T[i] = (_Float16)((k < 59) ? g4[k * 32 + n] : 0.f); }
    const float* g5 = w5g + (size_t)cell * 32 * 3;
    for (int i = tid; i < 16 * 32; i += 256) { int n = i >> 5, k = i & 31; w5T[i] = (_Float16)((n < 3) ? g5[k * 3 + n] : 0.f); }
    if (tid < 32) bias1[tid] = b1g[cell * 32 + tid];
    else if (tid < 80) { int n = tid - 32; bias2[n] = (n < 33) ? b2g[cell * 33 + n] : 0.f; }
    else if (tid < 112) bias3[tid - 80] = b3g[cell * 32 + (tid - 80)];
    else if (tid < 144) bias4[tid - 112] = b4g[cell * 32 + (tid - 112)];
    else if (tid < 160) { int n = tid - 144; bias5[n] = (n < 3) ? b5g[cell * 3 + n] : 0.f; }
  }
  __syncthreads();  // only uniform barrier; everything after is wave-private

  const int wave = tid >> 5, lane = tid & 31;
  _Float16* stag = stagAll + wave * 16 * 64;
  _Float16* edb = edAll + wave * 16 * 32;
  float* sig = sigAll + wave * 16;
  int* pmv = pmAll + wave * 16;
  int* mkv = mkAll + wave * 16;

  const int start = offsets[cell];
  const int npts = offsets[cell + 1] - start;
  const int nb = (npts + 15) >> 4;

  for (int b = wave; b < nb; b += 8) {
    const int m = lane & 15;
    const int gi = b * 16 + m;
    const int p = (gi < npts) ? pidx[start + gi] : -1;

    // ---- stage 0: encodings. lanes 0-15: positional (63); lanes 16-31: dir (27)
    if (lane < 16) {
      _Float16* row = stag + m * 64;
      if (p >= 0) {
        float x0 = xin[3 * p], x1 = xin[3 * p + 1], x2 = xin[3 * p + 2];
        int ok = (fabsf(x0) < 1.5f) && (fabsf(x1) < 1.5f) && (fabsf(x2) < 1.5f);
        pmv[m] = p; mkv[m] = ok;
        row[0] = (_Float16)x0; row[1] = (_Float16)x1; row[2] = (_Float16)x2;
        float f = 1.f;
#pragma unroll
        for (int j = 0; j < 10; ++j) {
          row[3 + 6 * j + 0] = (_Float16)__sinf(f * x0);
          row[3 + 6 * j + 1] = (_Float16)__sinf(f * x1);
          row[3 + 6 * j + 2] = (_Float16)__sinf(f * x2);
          row[3 + 6 * j + 3] = (_Float16)__cosf(f * x0);
          row[3 + 6 * j + 4] = (_Float16)__cosf(f * x1);
          row[3 + 6 * j + 5] = (_Float16)__cosf(f * x2);
          f *= 2.f;
        }
        row[63] = (_Float16)0.f;
      } else {
        pmv[m] = -1; mkv[m] = 0;
        for (int k = 0; k < 64; ++k) row[k] = (_Float16)0.f;
      }
    } else {
      _Float16* erow = edb + m * 32;
      if (p >= 0) {
        float d0 = din[3 * p], d1 = din[3 * p + 1], d2 = din[3 * p + 2];
        erow[0] = (_Float16)d0; erow[1] = (_Float16)d1; erow[2] = (_Float16)d2;
        float f = 1.f;
#pragma unroll
        for (int j = 0; j < 4; ++j) {
          erow[3 + 6 * j + 0] = (_Float16)__sinf(f * d0);
          erow[3 + 6 * j + 1] = (_Float16)__sinf(f * d1);
          erow[3 + 6 * j + 2] = (_Float16)__sinf(f * d2);
          erow[3 + 6 * j + 3] = (_Float16)__cosf(f * d0);
          erow[3 + 6 * j + 4] = (_Float16)__cosf(f * d1);
          erow[3 + 6 * j + 5] = (_Float16)__cosf(f * d2);
          f *= 2.f;
        }
        for (int k = 27; k < 32; ++k) erow[k] = (_Float16)0.f;
      } else {
        for (int k = 0; k < 32; ++k) erow[k] = (_Float16)0.f;
      }
    }
    WAVE_LDS_FENCE();

    const int n0 = lane & 15, mr = (lane >> 4) * 8;

    // ---- L1: 64 -> 32, ReLU
    v8f a1c0 = {}, a1c1 = {};
    {
      v16h a0 = load_a_frag(stag, lane, 0), a1 = load_a_frag(stag, lane, 1);
      a1c0 = WMMA(a0, load_b_frag(w1T, 64, lane, 0, 0), a1c0);
      a1c0 = WMMA(a1, load_b_frag(w1T, 64, lane, 1, 0), a1c0);
      a1c1 = WMMA(a0, load_b_frag(w1T, 64, lane, 0, 1), a1c1);
      a1c1 = WMMA(a1, load_b_frag(w1T, 64, lane, 1, 1), a1c1);
    }
#pragma unroll
    for (int r = 0; r < 8; ++r) {
      float v0 = a1c0[r] + bias1[n0];      v0 = v0 > 0.f ? v0 : 0.f;
      float v1 = a1c1[r] + bias1[16 + n0]; v1 = v1 > 0.f ? v1 : 0.f;
      stag[(mr + r) * 64 + n0] = (_Float16)v0;
      stag[(mr + r) * 64 + 16 + n0] = (_Float16)v1;
    }
    WAVE_LDS_FENCE();

    // ---- L2: 32 -> 33 (pad 48), ReLU; col0 = sigma, cols 1..32 -> staging 0..31
    v8f c0 = {}, c1 = {}, c2 = {};
    {
      v16h a = load_a_frag(stag, lane, 0);
      c0 = WMMA(a, load_b_frag(w2T, 32, lane, 0, 0), c0);
      c1 = WMMA(a, load_b_frag(w2T, 32, lane, 0, 1), c1);
      c2 = WMMA(a, load_b_frag(w2T, 32, lane, 0, 2), c2);
    }
#pragma unroll
    for (int r = 0; r < 8; ++r) {
      float v0 = c0[r] + bias2[n0];      v0 = v0 > 0.f ? v0 : 0.f;
      float v1 = c1[r] + bias2[16 + n0]; v1 = v1 > 0.f ? v1 : 0.f;
      float v2 = c2[r] + bias2[32 + n0]; v2 = v2 > 0.f ? v2 : 0.f;
      if (n0 == 0) { sig[mr + r] = v0; stag[(mr + r) * 64 + 31] = (_Float16)v2; }
      else stag[(mr + r) * 64 + (n0 - 1)] = (_Float16)v0;
      stag[(mr + r) * 64 + (15 + n0)] = (_Float16)v1;
    }
    WAVE_LDS_FENCE();

    // ---- L3: 32 -> 32 linear, then append dir encoding in cols 32..63
    v8f d0 = {}, d1 = {};
    {
      v16h a = load_a_frag(stag, lane, 0);
      d0 = WMMA(a, load_b_frag(w3T, 32, lane, 0, 0), d0);
      d1 = WMMA(a, load_b_frag(w3T, 32, lane, 0, 1), d1);
    }
#pragma unroll
    for (int r = 0; r < 8; ++r) {
      stag[(mr + r) * 64 + n0] = (_Float16)(d0[r] + bias3[n0]);
      stag[(mr + r) * 64 + 16 + n0] = (_Float16)(d1[r] + bias3[16 + n0]);
    }
    for (int i = lane; i < 16 * 16; i += 32) {  // 16 rows x 16 dwords of f16 pairs
      int rr = i >> 4, u = i & 15;
      ((unsigned*)(stag + rr * 64))[16 + u] = ((const unsigned*)(edb + rr * 32))[u];
    }
    WAVE_LDS_FENCE();

    // ---- L4: 64 -> 32, ReLU
    v8f e0 = {}, e1 = {};
    {
      v16h a0 = load_a_frag(stag, lane, 0), a1 = load_a_frag(stag, lane, 1);
      e0 = WMMA(a0, load_b_frag(w4T, 64, lane, 0, 0), e0);
      e0 = WMMA(a1, load_b_frag(w4T, 64, lane, 1, 0), e0);
      e1 = WMMA(a0, load_b_frag(w4T, 64, lane, 0, 1), e1);
      e1 = WMMA(a1, load_b_frag(w4T, 64, lane, 1, 1), e1);
    }
#pragma unroll
    for (int r = 0; r < 8; ++r) {
      float v0 = e0[r] + bias4[n0];      v0 = v0 > 0.f ? v0 : 0.f;
      float v1 = e1[r] + bias4[16 + n0]; v1 = v1 > 0.f ? v1 : 0.f;
      stag[(mr + r) * 64 + n0] = (_Float16)v0;
      stag[(mr + r) * 64 + 16 + n0] = (_Float16)v1;
    }
    WAVE_LDS_FENCE();

    // ---- L5: 32 -> 3, sigmoid, masked scatter of color + density
    v8f f0 = {};
    {
      v16h a = load_a_frag(stag, lane, 0);
      f0 = WMMA(a, load_b_frag(w5T, 32, lane, 0, 0), f0);
    }
#pragma unroll
    for (int r = 0; r < 8; ++r) {
      int row = mr + r, pp = pmv[row];
      if (n0 < 3) {
        float v = f0[r] + bias5[n0];
        v = 1.f / (1.f + __expf(-v));
        if (pp >= 0) outC[(size_t)pp * 3 + n0] = mkv[row] ? v : 0.f;
      } else if (n0 == 3) {
        if (pp >= 0) outD[pp] = mkv[row] ? sig[row] : 0.f;
      }
    }
    WAVE_LDS_FENCE();
  }
}

extern "C" void kernel_launch(void* const* d_in, const int* in_sizes, int n_in,
                              void* d_out, int out_size, void* d_ws, size_t ws_size,
                              hipStream_t stream) {
  const float* x  = (const float*)d_in[0];
  const float* dv = (const float*)d_in[1];
  const float* w1 = (const float*)d_in[2];  const float* b1 = (const float*)d_in[3];
  const float* w2 = (const float*)d_in[4];  const float* b2 = (const float*)d_in[5];
  const float* w3 = (const float*)d_in[6];  const float* b3 = (const float*)d_in[7];
  const float* w4 = (const float*)d_in[8];  const float* b4 = (const float*)d_in[9];
  const float* w5 = (const float*)d_in[10]; const float* b5 = (const float*)d_in[11];
  const int P = in_sizes[0] / 3;

  char* ws = (char*)d_ws;
  int* counts  = (int*)ws;
  int* offsets = (int*)(ws + (size_t)NCELLS * 4);
  int* cursor  = (int*)(ws + (size_t)NCELLS * 4 + (NCELLS + 1) * 4);
  int* pidx    = (int*)(ws + (size_t)NCELLS * 4 + (NCELLS + 1) * 4 + (size_t)NCELLS * 4);

  hipMemsetAsync(counts, 0, NCELLS * sizeof(int), stream);
  const int gb = (P + 255) / 256;
  count_kernel<<<gb, 256, 0, stream>>>(x, counts, P);
  scan_kernel<<<1, 1024, 0, stream>>>(counts, offsets, cursor);
  scatter_kernel<<<gb, 256, 0, stream>>>(x, cursor, pidx, P);

  float* outC = (float*)d_out;
  float* outD = outC + (size_t)P * 3;
  mlp_kernel<<<NCELLS, 256, 0, stream>>>(x, dv, w1, b1, w2, b2, w3, b3, w4, b4, w5, b5,
                                         offsets, pidx, outC, outD);
}
